// MultiHeadedAttention_5437428596926
// MI455X (gfx1250) — matrix operand order repaired and checked by reading, not compiled, not run on hardware
//
#include <hip/hip_runtime.h>
#include <hip/hip_bf16.h>
#include <stdint.h>

// ---------------------------------------------------------------------------
// MI455X (gfx1250) fused MHA: QKV projection (bf16 WMMA GEMM, double-buffered)
// + causal flash attention. B=4, S=2048, E=1024, H=16, DH=64
// ---------------------------------------------------------------------------

#define BB 4
#define SS 2048
#define EE 1024
#define HH 16
#define DH 64

typedef __attribute__((ext_vector_type(16))) __bf16 v16bf;
typedef __attribute__((ext_vector_type(8)))  float  v8f;
typedef __attribute__((ext_vector_type(2)))  __bf16 v2bf;

union Frag16 {            // 16 bf16 = 32 bytes = 8 VGPRs (one WMMA A/B operand)
    v16bf  v;
    __bf16 e[16];
    uint4  q[2];
};

__device__ __forceinline__ __bf16 f2bf(float f) { return (__bf16)f; }

// ===========================================================================
// Kernel 1: Q/K/V projection.
//   GEMM: [8192 x 1024] (x, f32) @ [1024 x 1024] (W reshaped, f32) -> bf16
//   Tiling: 256(M) x 64(N==DH) per block, K-chunks of 32, ping-pong LDS,
//   software-pipelined global->LDS staging. 8 waves x 32 rows, 8 WMMA/chunk.
//   grid = (M/256, H, 3{q,k,v}), block = 256
// ===========================================================================
__global__ void __launch_bounds__(256)
qkv_proj_kernel(const float* __restrict__ x,
                const float* __restrict__ Wq,
                const float* __restrict__ Wk,
                const float* __restrict__ Wv,
                __bf16* __restrict__ qb,
                __bf16* __restrict__ kb,
                __bf16* __restrict__ vb)
{
    __shared__ __bf16 Al[2][256 * 32];   // A tiles [m][k]   2 x 16 KB
    __shared__ __bf16 Bl[2][32 * 64];    // B tiles [k][n]   2 x  4 KB

    const int tid  = threadIdx.x;
    const int wave = tid >> 5;
    const int lane = tid & 31;
    const int m0   = blockIdx.x * 256;
    const int h    = blockIdx.y;             // N tile == head (N_TILE == DH)
    const int sel  = blockIdx.z;

    const float* W   = (sel == 0) ? Wq : (sel == 1) ? Wk : Wv;
    __bf16*      out = (sel == 0) ? qb : (sel == 1) ? kb : vb;

    // --- per-thread staging sources ---------------------------------------
    const float* xrow  = x + (size_t)(m0 + tid) * EE;                 // 1 row / thread
    const int    brow  = tid >> 3;                                    // 0..31
    const int    bpart = (tid & 7) * 8;                               // 0..56
    const float* wbase = W + (size_t)h * (EE * DH) + (size_t)brow * DH + bpart;

    float4 ar[8];        // 32 f32 of A row chunk
    float4 br[2];        // 8 f32 of B row chunk

    auto loadStage = [&](int k0) {
#pragma unroll
        for (int i = 0; i < 8; ++i)
            ar[i] = ((const float4*)(xrow + k0))[i];
        const float* bp = wbase + (size_t)k0 * DH;
        br[0] = ((const float4*)bp)[0];
        br[1] = ((const float4*)bp)[1];
    };
    auto storeStage = [&](int p) {
        union { __bf16 e[32]; uint4 q[4]; } ta;
#pragma unroll
        for (int i = 0; i < 8; ++i) {
            ta.e[4*i+0] = f2bf(ar[i].x); ta.e[4*i+1] = f2bf(ar[i].y);
            ta.e[4*i+2] = f2bf(ar[i].z); ta.e[4*i+3] = f2bf(ar[i].w);
        }
        uint4* dq = (uint4*)&Al[p][tid * 32];
        dq[0] = ta.q[0]; dq[1] = ta.q[1]; dq[2] = ta.q[2]; dq[3] = ta.q[3];

        union { __bf16 e[8]; uint4 q; } tb;
        tb.e[0]=f2bf(br[0].x); tb.e[1]=f2bf(br[0].y); tb.e[2]=f2bf(br[0].z); tb.e[3]=f2bf(br[0].w);
        tb.e[4]=f2bf(br[1].x); tb.e[5]=f2bf(br[1].y); tb.e[6]=f2bf(br[1].z); tb.e[7]=f2bf(br[1].w);
        *(uint4*)&Bl[p][brow * 64 + bpart] = tb.q;
    };

    v8f zero = {0.f,0.f,0.f,0.f,0.f,0.f,0.f,0.f};
    v8f c[2][4] = {{zero,zero,zero,zero},{zero,zero,zero,zero}};

    loadStage(0);
    storeStage(0);

    const int khalf = (lane < 16) ? 0 : 8;
    const int krow  = (lane & 15) + ((lane < 16) ? 0 : 16);

    for (int kc = 0; kc < EE / 32; ++kc) {
        __syncthreads();
        if (kc + 1 < EE / 32) loadStage((kc + 1) * 32);   // overlap with WMMAs

        const int p = kc & 1;
        Frag16 a0, a1;
        const __bf16* arow0 = &Al[p][(wave * 32 + (lane & 15)) * 32];
        const __bf16* arow1 = arow0 + 16 * 32;
        a0.q[0] = *(const uint4*)(arow0 + khalf);
        a0.q[1] = *(const uint4*)(arow0 + khalf + 16);
        a1.q[0] = *(const uint4*)(arow1 + khalf);
        a1.q[1] = *(const uint4*)(arow1 + khalf + 16);

#pragma unroll
        for (int j = 0; j < 4; ++j) {
            Frag16 bfr;
            const __bf16* bp = &Bl[p][khalf * 64 + j * 16 + (lane & 15)];
#pragma unroll
            for (int s = 0; s < 16; ++s)
                bfr.e[s] = bp[((s & 7) + ((s >> 3) << 4)) * 64];
            c[0][j] = __builtin_amdgcn_wmma_f32_16x16x32_bf16(
                false, a0.v, false, bfr.v, (short)0, c[0][j], false, false);
            c[1][j] = __builtin_amdgcn_wmma_f32_16x16x32_bf16(
                false, a1.v, false, bfr.v, (short)0, c[1][j], false, false);
        }
        if (kc + 1 < EE / 32) storeStage((kc + 1) & 1);   // waits loadcnt here
    }

    // --- epilogue: C layout (N=lane%16, M=vgpr(+8)) -> [B,H,S,DH] bf16 ----
#pragma unroll
    for (int part = 0; part < 2; ++part)
#pragma unroll
        for (int j = 0; j < 4; ++j)
#pragma unroll
            for (int v = 0; v < 8; ++v) {
                const int m = m0 + wave * 32 + part * 16 + ((lane < 16) ? v : v + 8);
                const int b = m >> 11;
                const int s = m & 2047;
                const int d = j * 16 + (lane & 15);
                out[(((size_t)b * HH + h) * SS + s) * DH + d] = f2bf(c[part][j][v]);
            }
}

// ===========================================================================
// Kernel 2: causal flash attention over bf16 Q/K/V, f32 accumulation.
//   grid = (B*H, S/128), block = 256 (8 waves, 16 query rows each)
//   64-key tiles, ping-pong K/V LDS buffers, pipelined staging,
//   16 WMMAs per tile per wave.
// ===========================================================================
__global__ void __launch_bounds__(256)
flash_attn_kernel(const __bf16* __restrict__ qb,
                  const __bf16* __restrict__ kb,
                  const __bf16* __restrict__ vb,
                  float* __restrict__ out)
{
    __shared__ __bf16 Kt[2][DH * 64];     // K tile transposed [d][t]  2 x 8 KB
    __shared__ __bf16 Vt[2][64 * DH];     // V tile row-major  [t][d]  2 x 8 KB
    __shared__ __bf16 Pl[8 * 16 * 64];    // per-wave P scratch            16 KB

    const int tid  = threadIdx.x;
    const int wave = tid >> 5;
    const int lane = tid & 31;
    const int bh   = blockIdx.x;
    const int q0   = blockIdx.y * 128;
    const int b    = bh >> 4;
    const int h    = bh & 15;

    const __bf16* Q  = qb + (size_t)bh * SS * DH;
    const __bf16* Kp = kb + (size_t)bh * SS * DH;
    const __bf16* Vp = vb + (size_t)bh * SS * DH;

    // --- load this wave's Q fragments once (2 chunks over DH=64) ----------
    Frag16 qa[2];
    {
        const int mrow  = q0 + wave * 16 + (lane & 15);
        const int khalf = (lane < 16) ? 0 : 8;
        const __bf16* qrow = Q + (size_t)mrow * DH;
#pragma unroll
        for (int cch = 0; cch < 2; ++cch) {
            qa[cch].q[0] = *(const uint4*)(qrow + cch * 32 + khalf);
            qa[cch].q[1] = *(const uint4*)(qrow + cch * 32 + khalf + 16);
        }
    }

    // --- staging: thread owns key-pair (2*r2, 2*r2+1), d range part..+7 ---
    const int r2   = tid >> 3;            // 0..31
    const int part = (tid & 7) * 8;       // 0..56
    uint4 kra, krb, vra, vrb;

    auto loadStage = [&](int k0) {
        const __bf16* ka = Kp + (size_t)(k0 + 2 * r2) * DH + part;
        const __bf16* va = Vp + (size_t)(k0 + 2 * r2) * DH + part;
        kra = *(const uint4*)ka;
        krb = *(const uint4*)(ka + DH);
        vra = *(const uint4*)va;
        vrb = *(const uint4*)(va + DH);
    };
    auto storeStage = [&](int p) {
        union { __bf16 e[8]; uint4 q; } ea, eb;
        ea.q = kra; eb.q = krb;
#pragma unroll
        for (int i = 0; i < 8; ++i) {     // transposed, key-pair packed b32
            union { v2bf v; __bf16 e[2]; } pr;
            pr.e[0] = ea.e[i]; pr.e[1] = eb.e[i];
            *(v2bf*)&Kt[p][(part + i) * 64 + 2 * r2] = pr.v;
        }
        *(uint4*)&Vt[p][(2 * r2)     * DH + part] = vra;
        *(uint4*)&Vt[p][(2 * r2 + 1) * DH + part] = vrb;
    };

    v8f zero = {0.f,0.f,0.f,0.f,0.f,0.f,0.f,0.f};
    v8f o[4] = {zero, zero, zero, zero};
    float mrun[8], lrun[8];
#pragma unroll
    for (int v = 0; v < 8; ++v) { mrun[v] = -1e30f; lrun[v] = 0.f; }

    const float scale  = 0.125f;               // 1/sqrt(64)
    const int   ntiles = ((q0 + 127) >> 6) + 1;

    loadStage(0);
    storeStage(0);

    for (int t = 0; t < ntiles; ++t) {
        __syncthreads();
        if (t + 1 < ntiles) {
            loadStage((t + 1) * 64);           // overlap with this tile's math
            if (t + 2 < ntiles) {              // L2 prefetch one more ahead
                __builtin_prefetch(Kp + (size_t)((t + 2) * 64 + 2 * r2) * DH + part, 0, 0);
                __builtin_prefetch(Vp + (size_t)((t + 2) * 64 + 2 * r2) * DH + part, 0, 0);
            }
        }
        const int p  = t & 1;
        const int k0 = t * 64;

        // --- S = Q @ K^T (2 d-chunks x 4 key-frags = 8 WMMAs) -------------
        v8f s[4] = {zero, zero, zero, zero};
#pragma unroll
        for (int cch = 0; cch < 2; ++cch) {
            const int dk = ((lane < 16) ? 0 : 8) + cch * 32;
#pragma unroll
            for (int j = 0; j < 4; ++j) {
                Frag16 bfr;
                const __bf16* bp = &Kt[p][dk * 64 + j * 16 + (lane & 15)];
#pragma unroll
                for (int s = 0; s < 16; ++s)
                    bfr.e[s] = bp[((s & 7) + ((s >> 3) << 4)) * 64];
                s[j] = __builtin_amdgcn_wmma_f32_16x16x32_bf16(
                    false, qa[cch].v, false, bfr.v, (short)0, s[j], false, false);
            }
        }

        // --- scale + causal mask ------------------------------------------
#pragma unroll
        for (int j = 0; j < 4; ++j)
#pragma unroll
            for (int v = 0; v < 8; ++v) {
                const int key = k0 + j * 16 + (lane & 15);
                const int row = q0 + wave * 16 + ((lane < 16) ? v : v + 8);
                float sc = s[j][v] * scale;
                s[j][v] = (key > row) ? -1e30f : sc;
            }

        // --- online softmax (row = C-frag vgpr slot, 16-lane halves) ------
#pragma unroll
        for (int v = 0; v < 8; ++v) {
            float mx = fmaxf(fmaxf(s[0][v], s[1][v]), fmaxf(s[2][v], s[3][v]));
#pragma unroll
            for (int off = 1; off < 16; off <<= 1)
                mx = fmaxf(mx, __shfl_xor(mx, off, 32));
            const float mnew = fmaxf(mrun[v], mx);
            const float corr = __expf(mrun[v] - mnew);
            mrun[v] = mnew;
            lrun[v] *= corr;
#pragma unroll
            for (int j = 0; j < 4; ++j) o[j][v] *= corr;
        }
#pragma unroll
        for (int j = 0; j < 4; ++j)
#pragma unroll
            for (int v = 0; v < 8; ++v)
                s[j][v] = __expf(s[j][v] - mrun[v]);
#pragma unroll
        for (int v = 0; v < 8; ++v) {
            float rs = (s[0][v] + s[1][v]) + (s[2][v] + s[3][v]);
#pragma unroll
            for (int off = 1; off < 16; off <<= 1)
                rs += __shfl_xor(rs, off, 32);
            lrun[v] += rs;
        }

        // --- P (C layout) -> per-wave LDS -> A layout ---------------------
        __bf16* pw = &Pl[wave * 16 * 64];
#pragma unroll
        for (int j = 0; j < 4; ++j)
#pragma unroll
            for (int v = 0; v < 8; ++v) {
                const int prow = (lane < 16) ? v : v + 8;
                const int pcol = j * 16 + (lane & 15);
                pw[prow * 64 + pcol] = f2bf(s[j][v]);
            }
        // same-wave LDS ops are in-order; region is wave-private -> no barrier

        // --- O += P @ V (2 key-chunks x 4 d-frags = 8 WMMAs) --------------
#pragma unroll
        for (int cch = 0; cch < 2; ++cch) {
            Frag16 pa;
            const int prw   = lane & 15;
            const int khalf = (lane < 16) ? 0 : 8;
            pa.q[0] = *(const uint4*)&pw[prw * 64 + cch * 32 + khalf];
            pa.q[1] = *(const uint4*)&pw[prw * 64 + cch * 32 + khalf + 16];
            const int kv = khalf + cch * 32;
#pragma unroll
            for (int j = 0; j < 4; ++j) {
                Frag16 vfr;
                const __bf16* vp = &Vt[p][kv * DH + j * 16 + (lane & 15)];
#pragma unroll
                for (int s = 0; s < 16; ++s)
                    vfr.e[s] = vp[((s & 7) + ((s >> 3) << 4)) * DH];
                o[j] = __builtin_amdgcn_wmma_f32_16x16x32_bf16(
                    false, pa.v, false, vfr.v, (short)0, o[j], false, false);
            }
        }

        if (t + 1 < ntiles) storeStage((t + 1) & 1);   // fill other buffer
    }

    // --- epilogue: O / l -> out[b][s][h*64+d] (f32) -----------------------
#pragma unroll
    for (int j = 0; j < 4; ++j)
#pragma unroll
        for (int v = 0; v < 8; ++v) {
            const int row = q0 + wave * 16 + ((lane < 16) ? v : v + 8);
            const int d   = j * 16 + (lane & 15);
            out[((size_t)b * SS + row) * (HH * DH) + h * DH + d] = o[j][v] / lrun[v];
        }
}

// ===========================================================================
extern "C" void kernel_launch(void* const* d_in, const int* in_sizes, int n_in,
                              void* d_out, int out_size, void* d_ws, size_t ws_size,
                              hipStream_t stream) {
    (void)in_sizes; (void)n_in; (void)out_size; (void)ws_size;

    const float* x  = (const float*)d_in[0];
    const float* Wq = (const float*)d_in[1];
    const float* Wk = (const float*)d_in[2];
    const float* Wv = (const float*)d_in[3];
    float* out = (float*)d_out;

    const size_t perMat = (size_t)BB * HH * SS * DH;   // 8 Mi elements
    __bf16* qbuf = (__bf16*)d_ws;
    __bf16* kbuf = qbuf + perMat;
    __bf16* vbuf = kbuf + perMat;

    dim3 g1((BB * SS) / 256, HH, 3);                   // 32 x 16 x 3
    qkv_proj_kernel<<<g1, 256, 0, stream>>>(x, Wq, Wk, Wv, qbuf, kbuf, vbuf);

    dim3 g2(BB * HH, SS / 128);                        // 64 x 16
    flash_attn_kernel<<<g2, 256, 0, stream>>>(qbuf, kbuf, vbuf, out);
}
